// HeteroGAT_11158325035015
// MI455X (gfx1250) — compile-verified, matrix operand order checked by reading, not run on hardware
//
#include <hip/hip_runtime.h>
#include <hip/hip_bf16.h>
#include <stdint.h>

// ---- problem sizes (from reference) ----
// NOTE: NP, NSN, NCN are all exact multiples of 16, so the WMMA GEMM needs no
// M-tail guard (100000 = 6250*16, 50000 = 3125*16, 64 = 4*16).
#define NP   100000
#define NSN  50000
#define NCN  64
#define ETO  300000
#define EHAS 200000
#define EFO  300000

typedef __attribute__((ext_vector_type(16))) __bf16 v16bf;
typedef __attribute__((ext_vector_type(8)))  float  v8f;

struct __align__(16) U4 { unsigned int w[4]; };

__device__ __forceinline__ unsigned short f2bf(float f) {
  unsigned u = __float_as_uint(f);
  u += 0x7FFFu + ((u >> 16) & 1u);          // round-to-nearest-even
  return (unsigned short)(u >> 16);
}
// order-preserving float<->uint encoding (for atomicMax over signed floats)
__device__ __forceinline__ unsigned fenc(float f) {
  unsigned u = __float_as_uint(f);
  return (u & 0x80000000u) ? ~u : (u | 0x80000000u);
}
__device__ __forceinline__ float fdec(unsigned u) {
  return (u & 0x80000000u) ? __uint_as_float(u & 0x7FFFFFFFu) : __uint_as_float(~u);
}

// ---------------- elementwise / init kernels ----------------
__global__ __launch_bounds__(256) void k_set_f32(float* p, float v, int n) {
  int t = blockIdx.x * 256 + threadIdx.x; if (t < n) p[t] = v;
}
__global__ __launch_bounds__(256) void k_set_u32(unsigned* p, unsigned v, int n) {
  int t = blockIdx.x * 256 + threadIdx.x; if (t < n) p[t] = v;
}
// weight f32[K x 128] -> bf16 packed into WMMA B-fragment order:
// Bp[(n_tile*(Kpad/32) + k_tile)*512 + lane*16 + j]
//   col  = n_tile*16 + (lane&15)
//   krow = k_tile*32 + ((lane>>4)<<3) + j + (j>=8 ? 8 : 0)
__global__ __launch_bounds__(256) void k_wpack_bf16(const float* __restrict__ s,
                                                    unsigned short* __restrict__ d,
                                                    int K, int Kpad) {
  int t = blockIdx.x * 256 + threadIdx.x;
  int total = Kpad * 128; if (t >= total) return;
  int j    = t & 15;
  int lane = (t >> 4) & 31;
  int rest = t >> 9;
  int kt   = Kpad >> 5;
  int kidx = rest % kt;
  int nidx = rest / kt;
  int col  = (nidx << 4) + (lane & 15);
  int krow = (kidx << 5) + ((lane >> 4) << 3) + j + ((j >= 8) ? 8 : 0);
  d[t] = (krow < K) ? f2bf(s[(size_t)krow * 128 + col]) : (unsigned short)0;
}
// feature f32[rows x K] -> bf16[rows x Kpad], zero-padded cols
__global__ __launch_bounds__(256) void k_feat_bf16(const float* __restrict__ s,
                                                   unsigned short* __restrict__ d,
                                                   int rows, int K, int Kpad) {
  int t = blockIdx.x * 256 + threadIdx.x;
  int total = rows * Kpad; if (t >= total) return;
  int r = t / Kpad, k = t - r * Kpad;
  d[t] = (k < K) ? f2bf(s[(size_t)r * K + k]) : (unsigned short)0;
}
// f32 -> bf16 (K == Kpad == 128), optional relu
__global__ __launch_bounds__(256) void k_cvt_bf16(const float* __restrict__ s,
                                                  unsigned short* __restrict__ d,
                                                  int total, int relu) {
  int t = blockIdx.x * 256 + threadIdx.x; if (t >= total) return;
  float v = s[t]; if (relu) v = fmaxf(v, 0.f);
  d[t] = f2bf(v);
}
// out[i,c] = sum of up to 4 biases[c]
__global__ __launch_bounds__(256) void k_bias_init(float* __restrict__ out, int total,
                                                   const float* b0, const float* b1,
                                                   const float* b2, const float* b3) {
  int t = blockIdx.x * 256 + threadIdx.x; if (t >= total) return;
  int c = t & 127;
  float v = b0[c];
  if (b1) v += b1[c];
  if (b2) v += b2[c];
  if (b3) v += b3[c];
  out[t] = v;
}

// ---------------- WMMA GEMM: C[M x 128] = A[M x Kpad](bf16) @ Bp(packed bf16) ----------------
// One wave per 16x16 tile; 8 waves/block cover the full N=128 strip of a 16-row tile.
// Requires M % 16 == 0 (true for all call sites), so the inner loop is pure:
// 2x b128 A-loads + 2x b128 packed-B loads + 1 v_wmma, no EXEC masking.
__global__ __launch_bounds__(256) void k_gemm(const unsigned short* __restrict__ A,
                                              const unsigned short* __restrict__ Bp,
                                              const float* __restrict__ bias,
                                              float* __restrict__ C,
                                              int Kpad, int relu) {
  const int lane   = threadIdx.x & 31;
  const int wave   = threadIdx.x >> 5;
  const int tile_m = blockIdx.x << 4;
  const int r      = lane & 15;
  const int koff   = (lane >> 4) << 3;     // 0 or 8 (ISA 16-bit fragment layout)
  const int row    = tile_m + r;
  const int col    = (wave << 4) + r;
  const int ktiles = Kpad >> 5;
  v8f acc = {};
  union Frag { v16bf v; unsigned short u[16]; U4 q[2]; };
  const unsigned short* ap = A + (size_t)row * Kpad + koff;
  const unsigned short* bp = Bp + (((size_t)wave * ktiles) << 9) + ((size_t)lane << 4);
  for (int kidx = 0; kidx < ktiles; ++kidx) {
    Frag a, b;
    a.q[0] = *(const U4*)(ap);
    a.q[1] = *(const U4*)(ap + 16);
    b.q[0] = *(const U4*)(bp);
    b.q[1] = *(const U4*)(bp + 8);
    ap += 32;
    bp += 512;
    acc = __builtin_amdgcn_wmma_f32_16x16x32_bf16(false, a.v, false, b.v,
                                                  (short)0, acc, false, false);
  }
  const int mb = tile_m + ((lane >> 4) << 3);   // C/D layout: lanes>=16 hold M+8
#pragma unroll
  for (int i = 0; i < 8; ++i) {
    float v = acc[i];
    if (bias) v += bias[col];
    if (relu) v = fmaxf(v, 0.f);
    C[(size_t)(mb + i) * 128 + col] = v;
  }
}

// ---------------- GAT kernels ----------------
// per-node per-head attention dot: out[i*4+h] = sum_c X[i,h*32+c]*att[h*32+c]
__global__ __launch_bounds__(256) void k_att_node(const float* __restrict__ X,
                                                  const float* __restrict__ att,
                                                  float* __restrict__ out, int n) {
  int t = blockIdx.x * 256 + threadIdx.x; if (t >= n * 4) return;
  int i = t >> 2, h = t & 3;
  const float* xp = X + (size_t)i * 128 + h * 32;
  const float* ap = att + h * 32;
  float s = 0.f;
#pragma unroll
  for (int c = 0; c < 32; ++c) s += xp[c] * ap[c];
  out[t] = s;
}
// per-head edge coefficient: we[h] = sum_c W_edge[h*32+c]*att_edge[h*32+c]
__global__ void k_edge_coef(const float* __restrict__ We, const float* __restrict__ ae,
                            float* __restrict__ we) {
  int h = threadIdx.x;
  if (h < 4) {
    float s = 0.f;
    for (int c = 0; c < 32; ++c) s += We[h * 32 + c] * ae[h * 32 + c];
    we[h] = s;
  }
}
// edge logits + leaky-relu + segment max (ordered-uint atomicMax)
__global__ __launch_bounds__(256) void k_edge_logit(const int* __restrict__ ei, int E,
                                                    const float* __restrict__ asn,
                                                    const float* __restrict__ adn,
                                                    const float* __restrict__ ea,
                                                    const float* __restrict__ we, int rev,
                                                    float* __restrict__ alog,
                                                    unsigned* __restrict__ m) {
  int t = blockIdx.x * 256 + threadIdx.x; if (t >= E * 4) return;
  int e = t >> 2, h = t & 3;
  int s = ei[e], d = ei[E + e];
  float a = asn[s * 4 + h] + adn[d * 4 + h];
  if (ea) { int ie = rev ? (E - 1 - e) : e; a += we[h] * ea[ie]; }
  a = (a > 0.f) ? a : 0.2f * a;               // leaky_relu(0.2)
  alog[t] = a;
  atomicMax(&m[d * 4 + h], fenc(a));
}
// e = exp(a - max); segment sum
__global__ __launch_bounds__(256) void k_edge_expsum(const int* __restrict__ ei, int E,
                                                     float* __restrict__ alog,
                                                     const unsigned* __restrict__ m,
                                                     float* __restrict__ sb) {
  int t = blockIdx.x * 256 + threadIdx.x; if (t >= E * 4) return;
  int e = t >> 2, h = t & 3;
  int d = ei[E + e];
  float v = expf(alog[t] - fdec(m[d * 4 + h]));
  alog[t] = v;
  atomicAdd(&sb[d * 4 + h], v);
}
// message scatter: out[dst] += alpha * S[src]; one thread = (edge, 4 channels)
__global__ __launch_bounds__(256) void k_edge_scatter(const int* __restrict__ ei, int E,
                                                      const float* __restrict__ alog,
                                                      const float* __restrict__ sb,
                                                      const float* __restrict__ S,
                                                      float* __restrict__ out) {
  int t = blockIdx.x * 256 + threadIdx.x;
  if (t >= E * 32) return;
  int e = t >> 5, q = t & 31;
  int c0 = q << 2, h = c0 >> 5;
  int s = ei[e], d = ei[E + e];
  float alpha = alog[(size_t)e * 4 + h] / (sb[d * 4 + h] + 1e-16f);
  const float* sp = S + (size_t)s * 128 + c0;
  float* op = out + (size_t)d * 128 + c0;
#pragma unroll
  for (int j = 0; j < 4; ++j) atomicAdd(&op[j], alpha * sp[j]);
}

// ---------------- GCN kernels ----------------
__global__ __launch_bounds__(256) void k_deg(const int* __restrict__ ei, int E,
                                             float* __restrict__ deg) {
  int t = blockIdx.x * 256 + threadIdx.x; if (t >= E) return;
  atomicAdd(&deg[ei[E + t]], 1.0f);
}
__global__ __launch_bounds__(256) void k_dinv(float* __restrict__ deg, int n) {
  int t = blockIdx.x * 256 + threadIdx.x; if (t >= n) return;
  deg[t] = rsqrtf(deg[t] + 1.0f);
}
__global__ __launch_bounds__(256) void k_gcn_self(const float* __restrict__ H,
                                                  const float* __restrict__ dinv,
                                                  float* __restrict__ out, int n) {
  int t = blockIdx.x * 256 + threadIdx.x; if (t >= n * 128) return;
  float di = dinv[t >> 7];
  out[t] += di * di * H[t];
}
__global__ __launch_bounds__(256) void k_gcn_scatter(const int* __restrict__ ei, int E,
                                                     const float* __restrict__ H,
                                                     const float* __restrict__ dinv,
                                                     float* __restrict__ out) {
  int t = blockIdx.x * 256 + threadIdx.x; if (t >= E * 32) return;
  int e = t >> 5, q = t & 31, c0 = q << 2;
  int s = ei[e], d = ei[E + e];
  float w = dinv[s] * dinv[d];
  const float* hp = H + (size_t)s * 128 + c0;
  float* op = out + (size_t)d * 128 + c0;
#pragma unroll
  for (int j = 0; j < 4; ++j) atomicAdd(&op[j], w * hp[j]);
}

// ---------------- host orchestration ----------------
typedef unsigned short ush;
struct GatW {
  const ush *Wsrc, *Wdst;
  const float *att_src, *att_dst, *bias;
  const float *Wedge, *att_edge;
};

extern "C" void kernel_launch(void* const* d_in, const int* in_sizes, int n_in,
                              void* d_out, int out_size, void* d_ws, size_t ws_size,
                              hipStream_t stream) {
  if (n_in < 112) return;
  auto F = [&](int i) { return (const float*)d_in[i]; };
  auto I = [&](int i) { return (const int*)d_in[i]; };
  auto cdiv = [](long long a, long long b) { return (int)((a + b - 1) / b); };

  char* cur = (char*)d_ws;
  auto alloc = [&](size_t bytes) -> void* {
    void* p = (void*)cur; cur += (bytes + 255) & ~(size_t)255; return p;
  };

  // bf16 node features
  ush* xpb = (ush*)alloc((size_t)NP * 160 * 2);
  ush* xsb = (ush*)alloc((size_t)NSN * 96 * 2);
  ush* xcb = (ush*)alloc((size_t)NCN * 32 * 2);
  ush* hpb = (ush*)alloc((size_t)NP * 128 * 2);
  ush* hsb = (ush*)alloc((size_t)NSN * 128 * 2);
  ush* hcb = (ush*)alloc((size_t)NCN * 128 * 2);
  ush* tb  = (ush*)alloc((size_t)NP * 128 * 2);
  // f32 scratch
  float* Sb   = (float*)alloc((size_t)NP * 128 * 4);
  float* Db   = (float*)alloc((size_t)NP * 128 * 4);
  float* outp = (float*)alloc((size_t)NP * 128 * 4);
  float* outs = (float*)alloc((size_t)NSN * 128 * 4);
  float* outc = (float*)alloc((size_t)NCN * 128 * 4);
  float* asn  = (float*)alloc((size_t)NP * 4 * 4);
  float* adn  = (float*)alloc((size_t)NP * 4 * 4);
  unsigned* mb = (unsigned*)alloc((size_t)NP * 4 * 4);
  float* sbuf = (float*)alloc((size_t)NP * 4 * 4);
  float* alog = (float*)alloc((size_t)ETO * 4 * 4);
  float* deg  = (float*)alloc((size_t)NP * 4);
  float* webuf = (float*)alloc(64);

  auto convW = [&](int idx, int K, int Kpad) -> const ush* {
    ush* p = (ush*)alloc((size_t)Kpad * 128 * 2);
    int total = Kpad * 128;
    k_wpack_bf16<<<cdiv(total, 256), 256, 0, stream>>>(F(idx), p, K, Kpad);
    return p;
  };
  auto gemm = [&](const ush* A, int M, int Kpad, const ush* B,
                  const float* bias, float* C, int relu) {
    k_gemm<<<dim3(M >> 4), 256, 0, stream>>>(A, B, bias, C, Kpad, relu);
  };

  // ---- convert+pack weights (pytree order: input_proj, l1..l3, linear_proj, shrink) ----
  const ush* Wip_c = convW(4, 7, 32);    const float* bip_c = F(5);
  const ush* Wip_p = convW(6, 138, 160); const float* bip_p = F(7);
  const ush* Wip_s = convW(8, 96, 96);   const float* bip_s = F(9);

  const ush *Wf[3], *Wfr[3];
  const float *bf_[3], *bfr[3];
  GatW has[3], hasr[3], to[3], tor[3];
  for (int l = 0; l < 3; ++l) {
    int L = 10 + 28 * l;
    Wf[l]  = convW(L + 0, 128, 128); bf_[l] = F(L + 1);
    Wfr[l] = convW(L + 2, 128, 128); bfr[l] = F(L + 3);
    has[l]  = { convW(L + 5, 128, 128),  convW(L + 4, 128, 128),
                F(L + 7),  F(L + 6),  F(L + 8),  nullptr, nullptr };
    hasr[l] = { convW(L + 10, 128, 128), convW(L + 9, 128, 128),
                F(L + 12), F(L + 11), F(L + 13), nullptr, nullptr };
    to[l]   = { convW(L + 16, 128, 128), convW(L + 14, 128, 128),
                F(L + 19), F(L + 17), F(L + 20), F(L + 15), F(L + 18) };
    tor[l]  = { convW(L + 23, 128, 128), convW(L + 21, 128, 128),
                F(L + 26), F(L + 24), F(L + 27), F(L + 22), F(L + 25) };
  }
  const ush* Wlp_c = convW(94, 128, 128);  const float* blp_c = F(95);
  const ush* Wlp_p = convW(96, 128, 128);  const float* blp_p = F(97);
  const ush* Wlp_s = convW(98, 128, 128);  const float* blp_s = F(99);
  const ush* Wsh_c = convW(100, 128, 128); const float* bsh_c = F(101);
  const ush* Wsh_p = convW(102, 128, 128); const float* bsh_p = F(103);
  const ush* Wsh_s = convW(104, 128, 128); const float* bsh_s = F(105);

  const int *eto = I(106), *etor = I(107), *ehas = I(108),
            *ehasr = I(109), *efo = I(110), *efor = I(111);
  const float* ea = F(3);

  // ---- input projections ----
  k_feat_bf16<<<cdiv((size_t)NP * 160, 256), 256, 0, stream>>>(F(0), xpb, NP, 138, 160);
  k_feat_bf16<<<cdiv((size_t)NSN * 96, 256), 256, 0, stream>>>(F(1), xsb, NSN, 96, 96);
  k_feat_bf16<<<cdiv((size_t)NCN * 32, 256), 256, 0, stream>>>(F(2), xcb, NCN, 7, 32);
  gemm(xpb, NP, 160, Wip_p, bip_p, Sb, 0);
  k_cvt_bf16<<<cdiv(NP * 128, 256), 256, 0, stream>>>(Sb, hpb, NP * 128, 0);
  gemm(xsb, NSN, 96, Wip_s, bip_s, Sb, 0);
  k_cvt_bf16<<<cdiv(NSN * 128, 256), 256, 0, stream>>>(Sb, hsb, NSN * 128, 0);
  gemm(xcb, NCN, 32, Wip_c, bip_c, Sb, 0);
  k_cvt_bf16<<<cdiv(NCN * 128, 256), 256, 0, stream>>>(Sb, hcb, NCN * 128, 0);

  auto runGat = [&](const ush* xsrc, int nsrc, const ush* xdst, int ndst,
                    const int* ei, int E, const GatW& w,
                    const float* eattr, int rev, float* out) {
    gemm(xsrc, nsrc, 128, w.Wsrc, nullptr, Sb, 0);
    gemm(xdst, ndst, 128, w.Wdst, nullptr, Db, 0);
    k_att_node<<<cdiv(nsrc * 4, 256), 256, 0, stream>>>(Sb, w.att_src, asn, nsrc);
    k_att_node<<<cdiv(ndst * 4, 256), 256, 0, stream>>>(Db, w.att_dst, adn, ndst);
    k_set_u32<<<cdiv(ndst * 4, 256), 256, 0, stream>>>(mb, 0x007FFFFFu, ndst * 4); // enc(-inf)
    k_set_f32<<<cdiv(ndst * 4, 256), 256, 0, stream>>>(sbuf, 0.f, ndst * 4);
    const float* we = nullptr;
    if (eattr) { k_edge_coef<<<1, 4, 0, stream>>>(w.Wedge, w.att_edge, webuf); we = webuf; }
    k_edge_logit<<<cdiv(E * 4, 256), 256, 0, stream>>>(ei, E, asn, adn, eattr, we, rev, alog, mb);
    k_edge_expsum<<<cdiv(E * 4, 256), 256, 0, stream>>>(ei, E, alog, mb, sbuf);
    k_edge_scatter<<<cdiv((long long)E * 32, 256), 256, 0, stream>>>(ei, E, alog, sbuf, Sb, out);
  };
  auto runGcn = [&](const ush* x, int n, const int* ei, int E, const ush* W, float* out) {
    gemm(x, n, 128, W, nullptr, Sb, 0);
    k_set_f32<<<cdiv(n, 256), 256, 0, stream>>>(deg, 0.f, n);
    k_deg<<<cdiv(E, 256), 256, 0, stream>>>(ei, E, deg);
    k_dinv<<<cdiv(n, 256), 256, 0, stream>>>(deg, n);
    k_gcn_self<<<cdiv(n * 128, 256), 256, 0, stream>>>(Sb, deg, out, n);
    k_gcn_scatter<<<cdiv((long long)E * 32, 256), 256, 0, stream>>>(ei, E, Sb, deg, out);
  };

  // ---- layers ----
  for (int l = 0; l < 3; ++l) {
    // out_sig = GAT(to): patients -> signatures, with edge attr
    k_bias_init<<<cdiv(NSN * 128, 256), 256, 0, stream>>>(outs, NSN * 128,
        to[l].bias, nullptr, nullptr, nullptr);
    runGat(hpb, NP, hsb, NSN, eto, ETO, to[l], ea, 0, outs);
    // out_pat = GAT(to_rev) + GAT(has_rev) + GCN(follows) + GCN(follows_rev)
    k_bias_init<<<cdiv(NP * 128, 256), 256, 0, stream>>>(outp, NP * 128,
        tor[l].bias, hasr[l].bias, bf_[l], bfr[l]);
    runGat(hsb, NSN, hpb, NP, etor, ETO, tor[l], ea, 1, outp);   // reversed edge attr
    runGat(hcb, NCN, hpb, NP, ehasr, EHAS, hasr[l], nullptr, 0, outp);
    runGcn(hpb, NP, efo, EFO, Wf[l], outp);
    runGcn(hpb, NP, efor, EFO, Wfr[l], outp);
    // out_cond = GAT(has): patients -> conditions
    k_bias_init<<<cdiv(NCN * 128, 256), 256, 0, stream>>>(outc, NCN * 128,
        has[l].bias, nullptr, nullptr, nullptr);
    runGat(hpb, NP, hcb, NCN, ehas, EHAS, has[l], nullptr, 0, outc);
    // relu -> bf16 for next layer
    k_cvt_bf16<<<cdiv(NP * 128, 256), 256, 0, stream>>>(outp, hpb, NP * 128, 1);
    k_cvt_bf16<<<cdiv(NSN * 128, 256), 256, 0, stream>>>(outs, hsb, NSN * 128, 1);
    k_cvt_bf16<<<cdiv(NCN * 128, 256), 256, 0, stream>>>(outc, hcb, NCN * 128, 1);
  }

  // ---- shrink (relu) + linear_proj -> d_out [patient | signature | condition] ----
  float* dout = (float*)d_out;
  gemm(hpb, NP, 128, Wsh_p, bsh_p, Sb, 1);
  k_cvt_bf16<<<cdiv(NP * 128, 256), 256, 0, stream>>>(Sb, tb, NP * 128, 0);
  gemm(tb, NP, 128, Wlp_p, blp_p, dout, 0);
  gemm(hsb, NSN, 128, Wsh_s, bsh_s, Sb, 1);
  k_cvt_bf16<<<cdiv(NSN * 128, 256), 256, 0, stream>>>(Sb, tb, NSN * 128, 0);
  gemm(tb, NSN, 128, Wlp_s, blp_s, dout + (size_t)NP * 128, 0);
  gemm(hcb, NCN, 128, Wsh_c, bsh_c, Sb, 1);
  k_cvt_bf16<<<cdiv(NCN * 128, 256), 256, 0, stream>>>(Sb, tb, NCN * 128, 0);
  gemm(tb, NCN, 128, Wlp_c, blp_c, dout + (size_t)(NP + NSN) * 128, 0);

  (void)in_sizes; (void)out_size; (void)ws_size;
}
